// CATAttention_89043261981441
// MI455X (gfx1250) — matrix operand order repaired and checked by reading, not compiled, not run on hardware
//
#include <hip/hip_runtime.h>
#include <cstdint>

#define S_LEN 2048
#define EMB   1024
#define NH    16
#define HD    64
#define BATCH 2
#define SCALE 0.125f   // HEAD_DIM^-0.5 = 64^-0.5

typedef __bf16 bf16_t;
typedef __attribute__((ext_vector_type(16))) __bf16 v16bf;
typedef __attribute__((ext_vector_type(8)))  __bf16 v8bf;
typedef __attribute__((ext_vector_type(8)))  float  v8f;
typedef __attribute__((ext_vector_type(4)))  unsigned int u32x4;
typedef __attribute__((ext_vector_type(8)))  int i32x8;
typedef __attribute__((ext_vector_type(4)))  int i32x4;

// ---------------------------------------------------------------------------
// WMMA helpers, following CDNA5 ISA VGPR layouts (CDNA5 ISA §7.12.2)
// ---------------------------------------------------------------------------

static __device__ __forceinline__ v8f wmma_bf16(v16bf a, v16bf b, v8f c) {
    // D = A(16x32) * B(32x16) + C, f32 accumulate
    return __builtin_amdgcn_wmma_f32_16x16x32_bf16(
        /*neg_a=*/false, a, /*neg_b=*/false, b,
        /*c_mod=*/(short)0, c, /*reuse_a=*/false, /*reuse_b=*/false);
}

// A-matrix 16x32 bf16 fragment from row-major [*, ld] source.
// Lane (lo = lane&15) holds row (row0+lo); VGPR p holds K pair
// k = hi*8 + (p<4 ? 2p : 16+2(p-4)).
static __device__ __forceinline__ v16bf load_a_frag(const bf16_t* __restrict__ A,
                                                    int ld, int row0, int k0) {
    int lane = threadIdx.x & 31;
    int lo = lane & 15, hi = lane >> 4;
    const bf16_t* rp = A + (size_t)(row0 + lo) * ld + k0 + hi * 8;
    v16bf a;
#pragma unroll
    for (int p = 0; p < 8; ++p) {
        int kb = (p < 4) ? (2 * p) : (16 + 2 * (p - 4));
        a[2 * p]     = rp[kb];
        a[2 * p + 1] = rp[kb + 1];
    }
    return a;
}

// B-matrix 32x16 bf16 fragment where B[k][n] = W[n][k] (W row-major, ld elems/row).
// Lane holds column n = n0 + (lane&15); K = k0 + hi*8 .. +7 then k0 + 16 + hi*8 .. +7
// -> two 16-byte vector loads per lane.
static __device__ __forceinline__ v16bf load_b_frag(const bf16_t* __restrict__ W,
                                                    int ld, int n0, int k0) {
    int lane = threadIdx.x & 31;
    int lo = lane & 15, hi = lane >> 4;
    const bf16_t* wp = W + (size_t)(n0 + lo) * ld + k0 + hi * 8;
    v8bf x0 = *(const v8bf*)wp;
    v8bf x1 = *(const v8bf*)(wp + 16);
    return __builtin_shufflevector(x0, x1, 0, 1, 2, 3, 4, 5, 6, 7, 8, 9, 10, 11, 12, 13, 14, 15);
}

// ---------------------------------------------------------------------------
// TDM: DMA a row of f32 from global memory into LDS (1 x nelem 2-D tile).
// Descriptor bitfields per CDNA5 ISA.
// Issued once per wave; caller gates to one wave per block and barriers.
// ---------------------------------------------------------------------------
static __device__ __forceinline__ void tdm_load_row_f32(const float* gsrc,
                                                        unsigned int lds_off,
                                                        int nelem) {
    uint64_t ga = (uint64_t)(uintptr_t)gsrc;
    u32x4 g0;
    g0[0] = 1u;                                             // count=1, user D#
    g0[1] = lds_off;                                        // lds_addr
    g0[2] = (unsigned int)ga;                               // global_addr[31:0]
    g0[3] = (unsigned int)((ga >> 32) & 0x1FFFFFFu)         // global_addr[56:32]
            | (2u << 30);                                   // type = 2 ("image")
    i32x8 g1;
    g1[0] = (2 << 16);                                      // data_size=4B, mask=0
    g1[1] = (nelem & 0xFFFF) << 16;                         // tensor_dim0[15:0]
    g1[2] = ((nelem >> 16) & 0xFFFF) | (1 << 16);           // dim0[31:16] | tensor_dim1=1
    g1[3] = (nelem & 0xFFFF) << 16;                         // tile_dim0 = nelem
    g1[4] = 1;                                              // tile_dim1=1, tile_dim2=0
    g1[5] = nelem;                                          // tensor_dim0_stride[31:0]
    g1[6] = 0;                                              // stride hi, dim1_stride lo
    g1[7] = 0;
    i32x4 g2 = {0, 0, 0, 0};
    i32x4 g3 = {0, 0, 0, 0};
#if __has_include(<hip/amd_detail/amd_gfx1250_TDM.h>)
    i32x8 g4 = {0, 0, 0, 0, 0, 0, 0, 0};                    // clang-23 6-arg form
    __builtin_amdgcn_tensor_load_to_lds(g0, g1, g2, g3, g4, 0);
#else
    __builtin_amdgcn_tensor_load_to_lds(g0, g1, g2, g3, 0); // ROCm 7.2 5-arg form
#endif
}

// ---------------------------------------------------------------------------
// Phase 0: f32 -> bf16 conversion
// ---------------------------------------------------------------------------
__global__ void k_cvt(const float* __restrict__ in, bf16_t* __restrict__ out, int n) {
    int i = (blockIdx.x * blockDim.x + threadIdx.x) * 4;
    if (i + 3 < n) {
        float4 v = *(const float4*)(in + i);
        out[i]     = (bf16_t)v.x;
        out[i + 1] = (bf16_t)v.y;
        out[i + 2] = (bf16_t)v.z;
        out[i + 3] = (bf16_t)v.w;
    } else {
        for (; i < n; ++i) out[i] = (bf16_t)in[i];
    }
}

// ---------------------------------------------------------------------------
// Phase 1: logits z[b,h,s] = (x[b,s,:] . W_A[h,:]) * SCALE   (stored [B,H,S])
// block 256 = 16 s-rows x 16 heads; grid = B*S/16
// ---------------------------------------------------------------------------
__global__ void k_logits(const float* __restrict__ x, const float* __restrict__ W_A,
                         float* __restrict__ wbuf) {
    int t = threadIdx.x;
    int h = t & 15, sl = t >> 4;
    int srow = blockIdx.x * 16 + sl;        // 0 .. B*S-1
    int b = srow / S_LEN, s = srow % S_LEN;
    const float4* xr = (const float4*)(x + (size_t)srow * EMB);
    const float4* wr = (const float4*)(W_A + (size_t)h * EMB);
    float acc = 0.f;
#pragma unroll 4
    for (int e = 0; e < EMB / 4; ++e) {
        float4 a = xr[e], w = wr[e];
        acc += a.x * w.x + a.y * w.y + a.z * w.z + a.w * w.w;
    }
    wbuf[((size_t)(b * NH + h)) * S_LEN + s] = acc * SCALE;
}

// ---------------------------------------------------------------------------
// Phase 2: softmax over s, in place on wbuf[B*H rows of length S]
// ---------------------------------------------------------------------------
__global__ void k_softmax(float* __restrict__ wbuf) {
    __shared__ float red[256];
    float* row = wbuf + (size_t)blockIdx.x * S_LEN;
    int t = threadIdx.x;

    float m = -3.4e38f;
    for (int i = t; i < S_LEN; i += 256) m = fmaxf(m, row[i]);
    red[t] = m; __syncthreads();
    for (int off = 128; off > 0; off >>= 1) {
        if (t < off) red[t] = fmaxf(red[t], red[t + off]);
        __syncthreads();
    }
    m = red[0]; __syncthreads();

    float sum = 0.f;
    for (int i = t; i < S_LEN; i += 256) {
        float e = __expf(row[i] - m);
        row[i] = e;
        sum += e;
    }
    red[t] = sum; __syncthreads();
    for (int off = 128; off > 0; off >>= 1) {
        if (t < off) red[t] += red[t + off];
        __syncthreads();
    }
    float inv = 1.f / red[0];
    for (int i = t; i < S_LEN; i += 256) row[i] *= inv;
}

// ---------------------------------------------------------------------------
// Phase 3: v = x . W_V^T  (M=B*S, N=EMB, K=EMB), stored transposed as
// vT[b,h,d,s] bf16 so the conv phase reads contiguous K(=s) runs.
// One wave computes a 16x64 strip; register double-buffered k-loop so the
// k+32 fragment loads overlap with the 4 WMMAs of chunk k.
// ---------------------------------------------------------------------------
__global__ void k_gemm_xv(const bf16_t* __restrict__ xbf, const bf16_t* __restrict__ wvbf,
                          bf16_t* __restrict__ vT) {
    int gw = (blockIdx.x * blockDim.x + threadIdx.x) >> 5;  // global wave id
    int mtile = gw >> 4;            // 256 row tiles
    int nstrip = gw & 15;           // 16 strips of 64 columns
    int row0 = mtile * 16;
    int n0 = nstrip * 64;

    v8f acc0 = {}, acc1 = {}, acc2 = {}, acc3 = {};
    v16bf a  = load_a_frag(xbf, EMB, row0, 0);
    v16bf b0 = load_b_frag(wvbf, EMB, n0,      0);
    v16bf b1 = load_b_frag(wvbf, EMB, n0 + 16, 0);
    v16bf b2 = load_b_frag(wvbf, EMB, n0 + 32, 0);
    v16bf b3 = load_b_frag(wvbf, EMB, n0 + 48, 0);
    for (int k0 = 0; k0 < EMB - 32; k0 += 32) {
        int kn = k0 + 32;
        v16bf an  = load_a_frag(xbf, EMB, row0, kn);
        v16bf bn0 = load_b_frag(wvbf, EMB, n0,      kn);
        v16bf bn1 = load_b_frag(wvbf, EMB, n0 + 16, kn);
        v16bf bn2 = load_b_frag(wvbf, EMB, n0 + 32, kn);
        v16bf bn3 = load_b_frag(wvbf, EMB, n0 + 48, kn);
        acc0 = wmma_bf16(a, b0, acc0);
        acc1 = wmma_bf16(a, b1, acc1);
        acc2 = wmma_bf16(a, b2, acc2);
        acc3 = wmma_bf16(a, b3, acc3);
        a = an; b0 = bn0; b1 = bn1; b2 = bn2; b3 = bn3;
    }
    acc0 = wmma_bf16(a, b0, acc0);
    acc1 = wmma_bf16(a, b1, acc1);
    acc2 = wmma_bf16(a, b2, acc2);
    acc3 = wmma_bf16(a, b3, acc3);

    int lane = threadIdx.x & 31, lo = lane & 15, hi = lane >> 4;
    int b = row0 / S_LEN;
    int s0 = (row0 % S_LEN) + hi * 8;   // lane's 8 rows are s0..s0+7 (contiguous)
    v8f accs[4] = {acc0, acc1, acc2, acc3};
#pragma unroll
    for (int tt = 0; tt < 4; ++tt) {
        int f = n0 + tt * 16 + lo;      // output feature
        int h = f >> 6, d = f & 63;
        bf16_t* dst = vT + (((size_t)(b * NH + h) * HD + d) * S_LEN) + s0;
        v8bf pk;
#pragma unroll
        for (int r = 0; r < 8; ++r) pk[r] = (bf16_t)accs[tt][r];
        *(v8bf*)dst = pk;               // 16B contiguous store
    }
}

// ---------------------------------------------------------------------------
// Phase 4: causal Toeplitz conv per (b,h):  out[i,d] = sum_{k<=i} w[i-k]*v[k,d]
// Softmax row staged into LDS by the Tensor Data Mover; A-fragment is the
// triangular Toeplitz tile built from LDS; B-fragment streams from vT with a
// one-chunk register pipeline. Fully-masked K-chunks are skipped.
// Block = 128 threads (4 waves), wave t owns d-tile t. Grid = B*H*(S/16).
// ---------------------------------------------------------------------------
__global__ void k_conv(const float* __restrict__ wbuf, const bf16_t* __restrict__ vT,
                       bf16_t* __restrict__ outbf) {
    __shared__ float wrow[S_LEN];
    int blk = blockIdx.x;
    int itile = blk & 127;              // S/16 = 128 row tiles
    int bh = blk >> 7;                  // 0..31
    int b = bh >> 4, h = bh & 15;
    int I0 = itile * 16;

    const float* wr = wbuf + (size_t)bh * S_LEN;
    if (threadIdx.x < 32) {             // one wave issues the TDM transfer
        tdm_load_row_f32(wr, (unsigned int)(uintptr_t)wrow, S_LEN);
        __builtin_amdgcn_s_wait_tensorcnt(0);
    }
    __syncthreads();

    int wave = threadIdx.x >> 5;
    int n0 = wave * 16;                 // d-tile base
    int lane = threadIdx.x & 31, lo = lane & 15, hi = lane >> 4;
    const bf16_t* vbase = vT + ((size_t)(b * NH + h) * HD) * S_LEN;

    v8f acc = {};
    int i_glob = I0 + lo;               // this lane's A-row
    int nk = I0 / 32 + 1;               // chunks covering k <= I0+15

    v16bf bcur = load_b_frag(vbase, S_LEN, n0, 0);
    for (int c = 0; c < nk - 1; ++c) {
        int K0 = c * 32;
        v16bf bnext = load_b_frag(vbase, S_LEN, n0, K0 + 32);
        v16bf a;
#pragma unroll
        for (int p = 0; p < 8; ++p) {
            int kb = K0 + hi * 8 + ((p < 4) ? (2 * p) : (16 + 2 * (p - 4)));
            int d0 = i_glob - kb;
            int d1 = d0 - 1;
            a[2 * p]     = (d0 >= 0) ? (bf16_t)wrow[d0] : (bf16_t)0.0f;
            a[2 * p + 1] = (d1 >= 0) ? (bf16_t)wrow[d1] : (bf16_t)0.0f;
        }
        acc = wmma_bf16(a, bcur, acc);
        bcur = bnext;
    }
    {   // final chunk (partially masked diagonal tile)
        int K0 = (nk - 1) * 32;
        v16bf a;
#pragma unroll
        for (int p = 0; p < 8; ++p) {
            int kb = K0 + hi * 8 + ((p < 4) ? (2 * p) : (16 + 2 * (p - 4)));
            int d0 = i_glob - kb;
            int d1 = d0 - 1;
            a[2 * p]     = (d0 >= 0) ? (bf16_t)wrow[d0] : (bf16_t)0.0f;
            a[2 * p + 1] = (d1 >= 0) ? (bf16_t)wrow[d1] : (bf16_t)0.0f;
        }
        acc = wmma_bf16(a, bcur, acc);
    }

    // out[b, s, h*64 + n0 + lo] for s = I0 + 8*hi + r
    int f = h * HD + n0 + lo;
#pragma unroll
    for (int r = 0; r < 8; ++r) {
        int s = I0 + 8 * hi + r;
        outbf[((size_t)(b * S_LEN + s)) * EMB + f] = (bf16_t)acc[r];
    }
}

// ---------------------------------------------------------------------------
// Phase 5: y = out . W_O^T + b_O  (f32 output), register double-buffered
// ---------------------------------------------------------------------------
__global__ void k_gemm_out(const bf16_t* __restrict__ outbf, const bf16_t* __restrict__ wobf,
                           const float* __restrict__ bO, float* __restrict__ y) {
    int gw = (blockIdx.x * blockDim.x + threadIdx.x) >> 5;
    int mtile = gw >> 4;
    int nstrip = gw & 15;
    int row0 = mtile * 16;
    int n0 = nstrip * 64;

    v8f acc0 = {}, acc1 = {}, acc2 = {}, acc3 = {};
    v16bf a  = load_a_frag(outbf, EMB, row0, 0);
    v16bf b0 = load_b_frag(wobf, EMB, n0,      0);
    v16bf b1 = load_b_frag(wobf, EMB, n0 + 16, 0);
    v16bf b2 = load_b_frag(wobf, EMB, n0 + 32, 0);
    v16bf b3 = load_b_frag(wobf, EMB, n0 + 48, 0);
    for (int k0 = 0; k0 < EMB - 32; k0 += 32) {
        int kn = k0 + 32;
        v16bf an  = load_a_frag(outbf, EMB, row0, kn);
        v16bf bn0 = load_b_frag(wobf, EMB, n0,      kn);
        v16bf bn1 = load_b_frag(wobf, EMB, n0 + 16, kn);
        v16bf bn2 = load_b_frag(wobf, EMB, n0 + 32, kn);
        v16bf bn3 = load_b_frag(wobf, EMB, n0 + 48, kn);
        acc0 = wmma_bf16(a, b0, acc0);
        acc1 = wmma_bf16(a, b1, acc1);
        acc2 = wmma_bf16(a, b2, acc2);
        acc3 = wmma_bf16(a, b3, acc3);
        a = an; b0 = bn0; b1 = bn1; b2 = bn2; b3 = bn3;
    }
    acc0 = wmma_bf16(a, b0, acc0);
    acc1 = wmma_bf16(a, b1, acc1);
    acc2 = wmma_bf16(a, b2, acc2);
    acc3 = wmma_bf16(a, b3, acc3);

    int lane = threadIdx.x & 31, lo = lane & 15, hi = lane >> 4;
    v8f accs[4] = {acc0, acc1, acc2, acc3};
#pragma unroll
    for (int tt = 0; tt < 4; ++tt) {
        int n = n0 + tt * 16 + lo;
        float bias = bO[n];
#pragma unroll
        for (int r = 0; r < 8; ++r) {
            int row = row0 + 8 * hi + r;
            y[(size_t)row * EMB + n] = accs[tt][r] + bias;
        }
    }
}

// ---------------------------------------------------------------------------
// Launch
// ---------------------------------------------------------------------------
extern "C" void kernel_launch(void* const* d_in, const int* in_sizes, int n_in,
                              void* d_out, int out_size, void* d_ws, size_t ws_size,
                              hipStream_t stream) {
    const float* x   = (const float*)d_in[0];   // [B,S,E]
    const float* W_A = (const float*)d_in[1];   // [H,E]
    const float* W_V = (const float*)d_in[2];   // [E,E]
    const float* W_O = (const float*)d_in[3];   // [E,E]
    const float* b_O = (const float*)d_in[4];   // [E]
    float* y = (float*)d_out;                   // [B,S,E] f32

    const size_t N_X  = (size_t)BATCH * S_LEN * EMB;   // 4194304
    const size_t N_W  = (size_t)EMB * EMB;             // 1048576
    const size_t N_WB = (size_t)BATCH * NH * S_LEN;    // 65536

    char* ws = (char*)d_ws;
    size_t off = 0;
    auto carve = [&](size_t bytes) -> void* {
        void* p = ws + off;
        off += (bytes + 255) & ~(size_t)255;
        return p;
    };
    bf16_t* xbf   = (bf16_t*)carve(N_X * 2);           // x in bf16
    bf16_t* wvbf  = (bf16_t*)carve(N_W * 2);           // W_V bf16
    bf16_t* wobf  = (bf16_t*)carve(N_W * 2);           // W_O bf16
    float*  wbuf  = (float*)carve(N_WB * 4);           // softmax weights [B,H,S]
    bf16_t* vT    = (bf16_t*)carve(N_X * 2);           // v transposed [B,H,D,S]
    bf16_t* outbf = (bf16_t*)carve(N_X * 2);           // conv output [B,S,E]

    // Phase 0: conversions
    k_cvt<<<(int)((N_X / 4 + 255) / 256), 256, 0, stream>>>(x, xbf, (int)N_X);
    k_cvt<<<(int)((N_W / 4 + 255) / 256), 256, 0, stream>>>(W_V, wvbf, (int)N_W);
    k_cvt<<<(int)((N_W / 4 + 255) / 256), 256, 0, stream>>>(W_O, wobf, (int)N_W);

    // Phase 1+2: logits + softmax
    k_logits<<<(BATCH * S_LEN) / 16, 256, 0, stream>>>(x, W_A, wbuf);
    k_softmax<<<BATCH * NH, 256, 0, stream>>>(wbuf);

    // Phase 3: v = x.W_V^T (WMMA) -> vT
    // waves = 256 row tiles * 16 n-strips = 4096; 8 waves/block -> 512 blocks
    k_gemm_xv<<<512, 256, 0, stream>>>(xbf, wvbf, vT);

    // Phase 4: causal Toeplitz conv (WMMA + TDM-staged weights) -> outbf
    k_conv<<<BATCH * NH * (S_LEN / 16), 128, 0, stream>>>(wbuf, vT, outbf);

    // Phase 5: y = out.W_O^T + b_O (WMMA)
    k_gemm_out<<<512, 256, 0, stream>>>(outbf, wobf, b_O, y);
}
